// QuantumLayer_33672543600855
// MI455X (gfx1250) — matrix operand order repaired and checked by reading, not compiled, not run on hardware
//
#include <hip/hip_runtime.h>

typedef __attribute__((ext_vector_type(16))) _Float16 v16h;
typedef __attribute__((ext_vector_type(8)))  float    v8f;

#define NTOK        65536   // B*S = 512*128
#define EDIM        512
#define NQ          4
#define WAVES_PB    8
#define TOK_PW      32      // tokens per wave (two 16-row WMMA tiles)

// ---- gather 16 floats in WMMA f16 A-fragment order for one lane ----
// lane<16 : K = colA+0..7 , colA+16..23   (colA = k*32)
// lane>=16: K = colA+0..7 , colA+16..23   (colA = k*32+8)
__device__ __forceinline__ void load_a16(const float* __restrict__ p, float* t) {
  float4 a0 = *(const float4*)(p);
  float4 a1 = *(const float4*)(p + 4);
  float4 a2 = *(const float4*)(p + 16);
  float4 a3 = *(const float4*)(p + 20);
  t[0]=a0.x;  t[1]=a0.y;  t[2]=a0.z;  t[3]=a0.w;
  t[4]=a1.x;  t[5]=a1.y;  t[6]=a1.z;  t[7]=a1.w;
  t[8]=a2.x;  t[9]=a2.y;  t[10]=a2.z; t[11]=a2.w;
  t[12]=a3.x; t[13]=a3.y; t[14]=a3.z; t[15]=a3.w;
}
// ---- 16 floats in WMMA f16 fragment order (B fragment: lane holds one column, K = p+0..7, p+16..23) ----
__device__ __forceinline__ void load_b16(const float* __restrict__ p, float m, float* t) {
  float4 b0 = *(const float4*)(p);
  float4 b1 = *(const float4*)(p + 4);
  float4 b2 = *(const float4*)(p + 16);
  float4 b3 = *(const float4*)(p + 20);
  t[0]=b0.x*m;  t[1]=b0.y*m;  t[2]=b0.z*m;  t[3]=b0.w*m;
  t[4]=b1.x*m;  t[5]=b1.y*m;  t[6]=b1.z*m;  t[7]=b1.w*m;
  t[8]=b2.x*m;  t[9]=b2.y*m;  t[10]=b2.z*m; t[11]=b2.w*m;
  t[12]=b3.x*m; t[13]=b3.y*m; t[14]=b3.z*m; t[15]=b3.w*m;
}
// ---- fp32 -> (hi,lo) f16 split: hi + lo ~= exact fp32 value ----
__device__ __forceinline__ void split16(const float* t, v16h& hi, v16h& lo) {
#pragma unroll
  for (int i = 0; i < 16; ++i) {
    _Float16 h = (_Float16)t[i];
    hi[i] = h;
    lo[i] = (_Float16)(t[i] - (float)h);
  }
}

__global__ __launch_bounds__(256) void qlayer_fused(
    const float* __restrict__ x,   // [NTOK, 512]
    const float* __restrict__ qw,  // [2, 8]
    const float* __restrict__ Wq,  // [4, 512]
    const float* __restrict__ bq,  // [4]
    const float* __restrict__ Wc,  // [512, 4]
    const float* __restrict__ bc,  // [512]
    float* __restrict__ out)       // [NTOK, 512]
{
  __shared__ float lqin [WAVES_PB][TOK_PW * 4];
  __shared__ float lqout[WAVES_PB][TOK_PW * 4];

  const int tid  = threadIdx.x;
  const int wv   = tid >> 5;
  const int lane = tid & 31;
  const int n    = lane & 15;   // A row within tile / B column
  const int half = lane >> 4;
  const int grpBase = (blockIdx.x * WAVES_PB + wv) * TOK_PW;

  // ================= GEMM1: q_in = x . Wq^T  (WMMA, split-f16 = fp32 acc) ====
  v8f acc0 = {};
  v8f acc1 = {};
  const float bmsk = (n < NQ) ? 1.0f : 0.0f;
  const float* xr0 = x + (size_t)(grpBase + n) * EDIM;        // tile0 rows
  const float* xr1 = xr0 + (size_t)16 * EDIM;                 // tile1 rows
  const float* wqr = Wq + (size_t)(n & 3) * EDIM;             // clamped row

#pragma unroll
  for (int k = 0; k < 16; ++k) {
    const int colA = k * 32 + half * 8;    // A-frag K runs: colA..+7, colA+16..+23
    const int colB = k * 32 + half * 8;    // B-frag K runs: colB..+7, colB+16..+23
    float ta0[16], ta1[16], tb[16];
    load_a16(xr0 + colA, ta0);
    load_a16(xr1 + colA, ta1);
    load_b16(wqr + colB, bmsk, tb);
    v16h A0h, A0l, A1h, A1l, Bh, Bl;
    split16(ta0, A0h, A0l);
    split16(ta1, A1h, A1l);
    split16(tb,  Bh,  Bl);
    // x*Wq = (Ah+Al)*(Bh+Bl) ~ Al*Bh + Ah*Bl + Ah*Bh  (lo*lo negligible)
    acc0 = __builtin_amdgcn_wmma_f32_16x16x32_f16(false, A0l, false, Bh, (short)0, acc0, false, false);
    acc0 = __builtin_amdgcn_wmma_f32_16x16x32_f16(false, A0h, false, Bl, (short)0, acc0, false, false);
    acc0 = __builtin_amdgcn_wmma_f32_16x16x32_f16(false, A0h, false, Bh, (short)0, acc0, false, false);
    acc1 = __builtin_amdgcn_wmma_f32_16x16x32_f16(false, A1l, false, Bh, (short)0, acc1, false, false);
    acc1 = __builtin_amdgcn_wmma_f32_16x16x32_f16(false, A1h, false, Bl, (short)0, acc1, false, false);
    acc1 = __builtin_amdgcn_wmma_f32_16x16x32_f16(false, A1h, false, Bh, (short)0, acc1, false, false);
  }

  // D layout: VGPR r, lane n(<16)/n+16 -> element (M = r + 8*half, N = n)
  if (n < NQ) {
#pragma unroll
    for (int r = 0; r < 8; ++r) {
      lqin[wv][(half * 8 + r) * 4 + n]      = acc0[r];
      lqin[wv][(16 + half * 8 + r) * 4 + n] = acc1[r];
    }
  }
  __syncthreads();

  // ================= quantum circuit: one token per lane =====================
  // shared weight trig (uniform -> scalar loads)
  float cR[2][4], sR[2][4], cY[2][4], sY[2][4];
#pragma unroll
  for (int l = 0; l < 2; ++l) {
#pragma unroll
    for (int i = 0; i < 4; ++i) {
      float a = 0.5f * qw[l * 8 + i];
      cR[l][i] = cosf(a);  sR[l][i] = sinf(a);
      float b = 0.5f * qw[l * 8 + 4 + i];
      cY[l][i] = cosf(b);  sY[l][i] = sinf(b);
    }
  }

  const float4 bqv = *(const float4*)bq;
  const float4 qv  = *(const float4*)&lqin[wv][lane * 4];
  float he[4] = { 0.5f * (qv.x + bqv.x), 0.5f * (qv.y + bqv.y),
                  0.5f * (qv.z + bqv.z), 0.5f * (qv.w + bqv.w) };
  float ce[4], se[4];
#pragma unroll
  for (int w = 0; w < 4; ++w) { ce[w] = cosf(he[w]); se[w] = sinf(he[w]); }

  // product state: amp_w = [cos, -i sin]; state[idx] = mag * (-i)^popcount(idx)
  float sr[16], si[16];
#pragma unroll
  for (int idx = 0; idx < 16; ++idx) {
    float m = 1.0f;
#pragma unroll
    for (int w = 0; w < 4; ++w)
      m *= ((idx >> (3 - w)) & 1) ? se[w] : ce[w];
    const int pc = ((idx >> 3) & 1) + ((idx >> 2) & 1) + ((idx >> 1) & 1) + (idx & 1);
    const int k4 = pc & 3;
    sr[idx] = (k4 == 0) ? m : ((k4 == 2) ? -m : 0.0f);
    si[idx] = (k4 == 1) ? -m : ((k4 == 3) ? m : 0.0f);
  }

#pragma unroll
  for (int l = 0; l < 2; ++l) {
    // ---- CRZ ring (diagonal phases) ----
#pragma unroll
    for (int i = 0; i < 4; ++i) {
      const int cb = 3 - i;                // control bit position (qubit i)
      const int tb = 3 - ((i + 1) & 3);    // target bit position
      const float cp = cR[l][i], sp = sR[l][i];
#pragma unroll
      for (int idx = 0; idx < 16; ++idx) {
        if ((idx >> cb) & 1) {
          const float pi_ = ((idx >> tb) & 1) ? sp : -sp;   // e^{i*±phi/2}
          const float r = sr[idx], q2 = si[idx];
          sr[idx] = r * cp - q2 * pi_;
          si[idx] = r * pi_ + q2 * cp;
        }
      }
    }
    // ---- RY on each qubit ----
#pragma unroll
    for (int i = 0; i < 4; ++i) {
      const int bp = 3 - i;
      const int stride = 1 << bp;
      const float ct = cY[l][i], st = sY[l][i];
#pragma unroll
      for (int idx = 0; idx < 16; ++idx) {
        if (!((idx >> bp) & 1)) {
          const int j2 = idx | stride;
          const float a  = sr[idx], b2  = sr[j2];
          sr[idx] = ct * a - st * b2;   sr[j2] = st * a + ct * b2;
          const float ai = si[idx], bi2 = si[j2];
          si[idx] = ct * ai - st * bi2; si[j2] = st * ai + ct * bi2;
        }
      }
    }
  }

  // probs -> PauliZ expvals
  float z0 = 0.f, z1 = 0.f, z2 = 0.f, z3 = 0.f;
#pragma unroll
  for (int idx = 0; idx < 16; ++idx) {
    const float p = sr[idx] * sr[idx] + si[idx] * si[idx];
    z0 += ((idx >> 3) & 1) ? -p : p;
    z1 += ((idx >> 2) & 1) ? -p : p;
    z2 += ((idx >> 1) & 1) ? -p : p;
    z3 += (idx & 1)        ? -p : p;
  }
  *(float4*)&lqout[wv][lane * 4] = make_float4(z0, z1, z2, z3);
  __syncthreads();

  // ================= GEMM2: out = q_out . Wc^T + bc (coalesced b128 stores) ==
#pragma unroll
  for (int j = 0; j < 4; ++j) {
    const int e0 = j * 128 + lane * 4;
    const float4 w0 = *(const float4*)(Wc + (size_t)(e0 + 0) * NQ);
    const float4 w1 = *(const float4*)(Wc + (size_t)(e0 + 1) * NQ);
    const float4 w2 = *(const float4*)(Wc + (size_t)(e0 + 2) * NQ);
    const float4 w3 = *(const float4*)(Wc + (size_t)(e0 + 3) * NQ);
    const float4 bv = *(const float4*)(bc + e0);
#pragma unroll 4
    for (int t = 0; t < TOK_PW; ++t) {
      const float4 qo = *(const float4*)&lqout[wv][t * 4];
      float4 o;
      o.x = bv.x + qo.x * w0.x + qo.y * w0.y + qo.z * w0.z + qo.w * w0.w;
      o.y = bv.y + qo.x * w1.x + qo.y * w1.y + qo.z * w1.z + qo.w * w1.w;
      o.z = bv.z + qo.x * w2.x + qo.y * w2.y + qo.z * w2.z + qo.w * w2.w;
      o.w = bv.w + qo.x * w3.x + qo.y * w3.y + qo.z * w3.z + qo.w * w3.w;
      *(float4*)(out + (size_t)(grpBase + t) * EDIM + e0) = o;
    }
  }
}

extern "C" void kernel_launch(void* const* d_in, const int* in_sizes, int n_in,
                              void* d_out, int out_size, void* d_ws, size_t ws_size,
                              hipStream_t stream) {
  const float* x  = (const float*)d_in[0];
  const float* qw = (const float*)d_in[1];
  const float* Wq = (const float*)d_in[2];
  const float* bq = (const float*)d_in[3];
  const float* Wc = (const float*)d_in[4];
  const float* bc = (const float*)d_in[5];
  float* out = (float*)d_out;

  dim3 grid(NTOK / (WAVES_PB * TOK_PW));  // 256 blocks
  dim3 block(WAVES_PB * 32);              // 256 threads = 8 wave32
  hipLaunchKernelGGL(qlayer_fused, grid, block, 0, stream,
                     x, qw, Wq, bq, Wc, bc, out);
}
